// Generator_38199439131224
// MI455X (gfx1250) — matrix operand order repaired and checked by reading, not compiled, not run on hardware
//
#include <hip/hip_runtime.h>
#include <hip/hip_bf16.h>
#include <math.h>
#include <stdint.h>

typedef __attribute__((ext_vector_type(16))) _Float16 v16h;
typedef __attribute__((ext_vector_type(8)))  float    v8f;
typedef __attribute__((ext_vector_type(4)))  unsigned int v4u;
typedef __attribute__((ext_vector_type(8)))  int      v8i;
typedef __attribute__((ext_vector_type(4)))  int      v4i;

#define LDH 132          // LDS row stride in halves (128 + 4 pad)
#define BN_EPS 1e-3f

#if __has_builtin(__builtin_amdgcn_tensor_load_to_lds) && __has_builtin(__builtin_amdgcn_s_wait_tensorcnt)
#define HAVE_TDM 1
#else
#define HAVE_TDM 0
#endif

// ------------------------- workspace layout (f16 half offsets) -------------------------
constexpr int H_W11  = 0;                      // 32x128 (K padded 2->32)
constexpr int H_W12  = H_W11  + 32 * 128;      // 32x32
constexpr int H_W13  = H_W12  + 32 * 32;       // 32x32
constexpr int H_W14  = H_W13  + 32 * 32;       // 32x128
constexpr int H_R1K  = H_W14  + 32 * 128;      // 128x128
constexpr int H_R1Q  = H_R1K  + 128 * 128;
constexpr int H_R1V  = H_R1Q  + 128 * 128;
constexpr int H_R1P  = H_R1V  + 128 * 128;
constexpr int H_R2K  = H_R1P  + 128 * 128;
constexpr int H_R2Q  = H_R2K  + 128 * 128;
constexpr int H_R2V  = H_R2Q  + 128 * 128;
constexpr int H_R2P  = H_R2V  + 128 * 128;
constexpr int H_W21  = H_R2P  + 128 * 128;     // 128x128
constexpr int H_W22  = H_W21  + 128 * 128;     // 128x32
constexpr int H_W23  = H_W22  + 128 * 32;      // 32x32
constexpr int H_W24  = H_W23  + 32 * 32;       // 32x128
constexpr int H_WGEO = H_W24  + 32 * 128;      // 128x16 (N padded 2->16)
constexpr int H_TOTAL = H_WGEO + 128 * 16;     // 168960 halves

// ------------------------- workspace layout (f32 offsets, after f16 region) ------------
constexpr int F_B11  = 0;
constexpr int F_B12  = F_B11 + 128;
constexpr int F_B13  = F_B12 + 32;
constexpr int F_B14  = F_B13 + 32;
constexpr int F_R1BK = F_B14 + 128;
constexpr int F_R1BQ = F_R1BK + 128;
constexpr int F_R1BV = F_R1BQ + 128;
constexpr int F_R1BP = F_R1BV + 128;
constexpr int F_S2   = F_R1BP + 128;
constexpr int F_T2   = F_S2 + 128;
constexpr int F_R2BK = F_T2 + 128;
constexpr int F_R2BQ = F_R2BK + 128;
constexpr int F_R2BV = F_R2BQ + 128;
constexpr int F_R2BP = F_R2BV + 128;
constexpr int F_S4   = F_R2BP + 128;
constexpr int F_T4   = F_S4 + 128;
constexpr int F_B21  = F_T4 + 128;
constexpr int F_B22  = F_B21 + 128;
constexpr int F_B23  = F_B22 + 32;
constexpr int F_B24  = F_B23 + 32;
constexpr int F_BGEO = F_B24 + 128;
constexpr int F_TOTAL = F_BGEO + 16;           // 2192 floats

struct Ptrs { const float* p[83]; };

// ======================= prep: BN-fold + f16 convert (runs once per call) ==============
__device__ void fold_w(const float* W, int K, int N, int Kp, int Np,
                       const float* g, const float* bb, const float* m, const float* v,
                       float es, _Float16* Wf, int tid, int nt) {
  for (int i = tid; i < Kp * Np; i += nt) {
    int k = i / Np, n = i % Np;
    float w = 0.0f;
    if (k < K && n < N) {
      float s = 1.0f;
      if (g) s = g[n] * rsqrtf(v[n] + BN_EPS);
      w = W[k * N + n] * s * es;
    }
    Wf[i] = (_Float16)w;
  }
}

__device__ void fold_b(const float* cb, int N, int Np,
                       const float* g, const float* bb, const float* m, const float* v,
                       float es, float* bf, int tid, int nt) {
  for (int n = tid; n < Np; n += nt) {
    float r = 0.0f;
    if (n < N) {
      float s = 1.0f, t = 0.0f;
      if (g) { s = g[n] * rsqrtf(v[n] + BN_EPS); t = bb[n] - m[n] * s; }
      r = cb[n] * es * s + t;
    }
    bf[n] = r;
  }
}

__device__ void bn_st(const float* g, const float* bb, const float* m, const float* v,
                      float* s, float* t, int N, int tid, int nt) {
  for (int n = tid; n < N; n += nt) {
    float sv = g[n] * rsqrtf(v[n] + BN_EPS);
    s[n] = sv;
    t[n] = bb[n] - m[n] * sv;
  }
}

__global__ void pcn_prep(Ptrs P, _Float16* __restrict__ wh, float* __restrict__ wf) {
  const int tid = threadIdx.x, nt = blockDim.x;
  const float QS = 0.08838834764831845f;  // 1/sqrt(128), folded into query conv
  fold_w(P.p[49],  2, 128, 32, 128, P.p[1],  P.p[0],  P.p[2],  P.p[3],  1.f, wh + H_W11, tid, nt);
  fold_b(P.p[48], 128, 128,      P.p[1],  P.p[0],  P.p[2],  P.p[3],  1.f, wf + F_B11, tid, nt);
  fold_w(P.p[51],  2,  32, 32,  32, P.p[5],  P.p[4],  P.p[6],  P.p[7],  1.f, wh + H_W12, tid, nt);
  fold_b(P.p[50],  32,  32,      P.p[5],  P.p[4],  P.p[6],  P.p[7],  1.f, wf + F_B12, tid, nt);
  fold_w(P.p[53], 32,  32, 32,  32, P.p[9],  P.p[8],  P.p[10], P.p[11], 1.f, wh + H_W13, tid, nt);
  fold_b(P.p[52],  32,  32,      P.p[9],  P.p[8],  P.p[10], P.p[11], 1.f, wf + F_B13, tid, nt);
  fold_w(P.p[55], 32, 128, 32, 128, P.p[13], P.p[12], P.p[14], P.p[15], 1.f, wh + H_W14, tid, nt);
  fold_b(P.p[54], 128, 128,      P.p[13], P.p[12], P.p[14], P.p[15], 1.f, wf + F_B14, tid, nt);
  fold_w(P.p[67], 128, 128, 128, 128, nullptr, nullptr, nullptr, nullptr, 1.f, wh + H_R1K, tid, nt);
  fold_b(P.p[66], 128, 128, nullptr, nullptr, nullptr, nullptr, 1.f, wf + F_R1BK, tid, nt);
  fold_w(P.p[71], 128, 128, 128, 128, nullptr, nullptr, nullptr, nullptr, QS,  wh + H_R1Q, tid, nt);
  fold_b(P.p[70], 128, 128, nullptr, nullptr, nullptr, nullptr, QS,  wf + F_R1BQ, tid, nt);
  fold_w(P.p[73], 128, 128, 128, 128, nullptr, nullptr, nullptr, nullptr, 1.f, wh + H_R1V, tid, nt);
  fold_b(P.p[72], 128, 128, nullptr, nullptr, nullptr, nullptr, 1.f, wf + F_R1BV, tid, nt);
  fold_w(P.p[69], 128, 128, 128, 128, P.p[33], P.p[32], P.p[34], P.p[35], 1.f, wh + H_R1P, tid, nt);
  fold_b(P.p[68], 128, 128,        P.p[33], P.p[32], P.p[34], P.p[35], 1.f, wf + F_R1BP, tid, nt);
  bn_st(P.p[37], P.p[36], P.p[38], P.p[39], wf + F_S2, wf + F_T2, 128, tid, nt);
  fold_w(P.p[75], 128, 128, 128, 128, nullptr, nullptr, nullptr, nullptr, 1.f, wh + H_R2K, tid, nt);
  fold_b(P.p[74], 128, 128, nullptr, nullptr, nullptr, nullptr, 1.f, wf + F_R2BK, tid, nt);
  fold_w(P.p[79], 128, 128, 128, 128, nullptr, nullptr, nullptr, nullptr, QS,  wh + H_R2Q, tid, nt);
  fold_b(P.p[78], 128, 128, nullptr, nullptr, nullptr, nullptr, QS,  wf + F_R2BQ, tid, nt);
  fold_w(P.p[81], 128, 128, 128, 128, nullptr, nullptr, nullptr, nullptr, 1.f, wh + H_R2V, tid, nt);
  fold_b(P.p[80], 128, 128, nullptr, nullptr, nullptr, nullptr, 1.f, wf + F_R2BV, tid, nt);
  fold_w(P.p[77], 128, 128, 128, 128, P.p[41], P.p[40], P.p[42], P.p[43], 1.f, wh + H_R2P, tid, nt);
  fold_b(P.p[76], 128, 128,        P.p[41], P.p[40], P.p[42], P.p[43], 1.f, wf + F_R2BP, tid, nt);
  bn_st(P.p[45], P.p[44], P.p[46], P.p[47], wf + F_S4, wf + F_T4, 128, tid, nt);
  fold_w(P.p[57], 128, 128, 128, 128, P.p[17], P.p[16], P.p[18], P.p[19], 1.f, wh + H_W21, tid, nt);
  fold_b(P.p[56], 128, 128,        P.p[17], P.p[16], P.p[18], P.p[19], 1.f, wf + F_B21, tid, nt);
  fold_w(P.p[59], 128,  32, 128,  32, P.p[21], P.p[20], P.p[22], P.p[23], 1.f, wh + H_W22, tid, nt);
  fold_b(P.p[58],  32,  32,        P.p[21], P.p[20], P.p[22], P.p[23], 1.f, wf + F_B22, tid, nt);
  fold_w(P.p[61], 32,  32, 32,  32,  P.p[25], P.p[24], P.p[26], P.p[27], 1.f, wh + H_W23, tid, nt);
  fold_b(P.p[60],  32,  32,        P.p[25], P.p[24], P.p[26], P.p[27], 1.f, wf + F_B23, tid, nt);
  fold_w(P.p[63], 32, 128, 32, 128,  P.p[29], P.p[28], P.p[30], P.p[31], 1.f, wh + H_W24, tid, nt);
  fold_b(P.p[62], 128, 128,        P.p[29], P.p[28], P.p[30], P.p[31], 1.f, wf + F_B24, tid, nt);
  fold_w(P.p[65], 128, 2, 128, 16, nullptr, nullptr, nullptr, nullptr, 1.f, wh + H_WGEO, tid, nt);
  fold_b(P.p[64],   2, 16, nullptr, nullptr, nullptr, nullptr, 1.f, wf + F_BGEO, tid, nt);
}

// ======================= TDM: global weight block -> LDS (D# per ISA 08 §8) ============
#if HAVE_TDM
__device__ __forceinline__ void tdm_load_weights(const _Float16* src, int K, int N,
                                                 _Float16* lds_dst) {
  const unsigned long long ga = (unsigned long long)(uintptr_t)src;
  const unsigned int       la = (unsigned int)(uintptr_t)lds_dst;  // low 32b = LDS byte addr
  v4u g0;
  g0[0] = 1u;                                        // count=1, user descriptor
  g0[1] = la;                                        // lds_addr
  g0[2] = (unsigned int)ga;                          // global_addr[31:0]
  g0[3] = ((unsigned int)(ga >> 32) & 0x01FFFFFFu)   // global_addr[56:32]
          | 0x80000000u;                             // type=2 ("image")
  v8i g1;
  g1[0] = 0x00010000;                                // data_size=1 (2 bytes), mask=0
  g1[1] = (int)((N & 0xFFFF) << 16);                 // tensor_dim0[15:0]
  g1[2] = (int)(((N >> 16) & 0xFFFF) | ((K & 0xFFFF) << 16));  // dim0 hi | tensor_dim1 lo
  g1[3] = (int)(((K >> 16) & 0xFFFF) | ((N & 0xFFFF) << 16));  // dim1 hi | tile_dim0
  g1[4] = (int)(K & 0xFFFF);                         // tile_dim1 (tile_dim2 = 0)
  g1[5] = (int)N;                                    // tensor_dim0_stride[31:0]
  g1[6] = 0;                                         // stride0 hi | stride1 lo
  g1[7] = 0;
  v4i z4 = {0, 0, 0, 0};
#if defined(__clang_major__) && __clang_major__ >= 23
  v8i z8 = {0, 0, 0, 0, 0, 0, 0, 0};
  __builtin_amdgcn_tensor_load_to_lds(g0, g1, z4, z4, z8, 0);
#else
  __builtin_amdgcn_tensor_load_to_lds(g0, g1, z4, z4, 0);
#endif
}
#endif

// issue next-stage weight DMA (wave 0 only); overlaps with current GEMM
__device__ __forceinline__ void stage_load(const _Float16* src, int K, int N,
                                           _Float16* dst, int tid, int wave) {
#if HAVE_TDM
  if (wave == 0) tdm_load_weights(src, K, N, dst);
#else
  const int cnt = K * N;
  for (int i = tid; i < cnt; i += 256) dst[i] = src[i];
#endif
}

__device__ __forceinline__ void stage_sync(int wave) {
#if HAVE_TDM
  if (wave == 0) __builtin_amdgcn_s_wait_tensorcnt((short)0);
#endif
  __syncthreads();
}

// ======================= WMMA fragment loaders (ISA 7.12.2 layouts) ====================
__device__ __forceinline__ v16h load_a_frag(const _Float16* A, int lda, int m0, int k0, int lane) {
  const int m  = m0 + (lane & 15);
  const int kb = (lane >> 4) << 3;
  const _Float16* p = A + m * lda + k0;
  v16h r;
#pragma unroll
  for (int h = 0; h < 8; ++h) r[h] = p[kb + h];
#pragma unroll
  for (int h = 0; h < 8; ++h) r[8 + h] = p[16 + kb + h];
  return r;
}

__device__ __forceinline__ v16h load_b_frag(const _Float16* B, int ldb, int k0, int n0, int lane) {
  const int kb = (lane >> 4) << 3;
  const _Float16* p = B + k0 * ldb + n0 + (lane & 15);
  v16h r;
#pragma unroll
  for (int h = 0; h < 8; ++h) r[h] = p[(kb + h) * ldb];
#pragma unroll
  for (int h = 0; h < 8; ++h) r[8 + h] = p[(16 + kb + h) * ldb];
  return r;
}

__device__ __forceinline__ v16h load_bt_frag(const _Float16* S, int lds_, int k0, int n0, int lane) {
  const int kb = (lane >> 4) << 3;
  const _Float16* p = S + (n0 + (lane & 15)) * lds_ + k0;
  v16h r;
#pragma unroll
  for (int h = 0; h < 8; ++h) r[h] = p[kb + h];
#pragma unroll
  for (int h = 0; h < 8; ++h) r[8 + h] = p[16 + kb + h];
  return r;
}

__device__ __forceinline__ void epilogue_tile(
    v8f acc, _Float16* dst, int ldd, int m0, int n0, int lane,
    const float* bias, bool inner_relu,
    const _Float16* res, int ldr,
    const float* sc, const float* sh, bool relu_out) {
  const int n  = n0 + (lane & 15);
  const int mb = m0 + ((lane >> 4) << 3);
  const float bb = bias ? bias[n] : 0.0f;
#pragma unroll
  for (int j = 0; j < 8; ++j) {
    float d = acc[j] + bb;
    if (inner_relu) d = fmaxf(d, 0.0f);
    if (res) d += (float)res[(mb + j) * ldr + n];
    if (sc)  d = d * sc[n] + sh[n];
    if (relu_out) d = fmaxf(d, 0.0f);
    dst[(mb + j) * ldd + n] = (_Float16)d;
  }
}

template<bool BT>
__device__ __forceinline__ void gemm_stage(
    const _Float16* A, int lda, const _Float16* B, int ldb,
    int N, int K, _Float16* dst, int ldd,
    const float* bias, bool inner_relu,
    const _Float16* res, int ldr,
    const float* sc, const float* sh, bool relu_out,
    int wave, int lane) {
  const int ntiles = 8 * (N >> 4);
  for (int t = wave; t < ntiles; t += 8) {
    const int m0 = (t & 7) << 4;
    const int n0 = (t >> 3) << 4;
    v8f acc = {};
    for (int k0 = 0; k0 < K; k0 += 32) {
      v16h a = load_a_frag(A, lda, m0, k0, lane);
      v16h b = BT ? load_bt_frag(B, ldb, k0, n0, lane)
                  : load_b_frag(B, ldb, k0, n0, lane);
      acc = __builtin_amdgcn_wmma_f32_16x16x32_f16(false, a, false, b, (short)0, acc,
                                                   false, false);
    }
    epilogue_tile(acc, dst, ldd, m0, n0, lane, bias, inner_relu, res, ldr, sc, sh, relu_out);
  }
}

// relation block: weights consumed from LDS (w0/w1 double buffer), next stage prefetched
__device__ void relation_block(_Float16* Xe, _Float16* Kb, _Float16* Qb, _Float16* Ab,
                               _Float16* w0, _Float16* w1, const float* Cf,
                               const _Float16* wh, int oQ, int oV, int oP,
                               int obK, int obQ, int obV, int obP, int oS, int oT,
                               const _Float16* nextW, int nK, int nN,
                               int tid, int wave, int lane) {
  // k = X@Wk + bk        (weights already in w0; prefetch q -> w1)
  stage_load(wh + oQ, 128, 128, w1, tid, wave);
  gemm_stage<false>(Xe, LDH, w0, 128, 128, 128, Kb, LDH, Cf + obK, false, nullptr, 0, nullptr, nullptr, false, wave, lane);
  stage_sync(wave);
  // q' = (X@Wq + bq)/sqrt(n)  (scale folded)
  gemm_stage<false>(Xe, LDH, w1, 128, 128, 128, Qb, LDH, Cf + obQ, false, nullptr, 0, nullptr, nullptr, false, wave, lane);
  __syncthreads();
  // at[m,n] = q'[m].k[n]   (prefetch v -> w0 overlaps the attention GEMM)
  stage_load(wh + oV, 128, 128, w0, tid, wave);
  gemm_stage<true>(Qb, LDH, Kb, LDH, 128, 128, Ab, LDH, nullptr, false, nullptr, 0, nullptr, nullptr, false, wave, lane);
  stage_sync(wave);
  // v = X@Wv + bv -> Qb (reuse); prefetch proj -> w1
  stage_load(wh + oP, 128, 128, w1, tid, wave);
  gemm_stage<false>(Xe, LDH, w0, 128, 128, 128, Qb, LDH, Cf + obV, false, nullptr, 0, nullptr, nullptr, false, wave, lane);
  stage_sync(wave);
  // o = at @ v -> Kb; prefetch the block after this relation -> w0
  stage_load(nextW, nK, nN, w0, tid, wave);
  gemm_stage<false>(Ab, LDH, Qb, LDH, 128, 128, Kb, LDH, nullptr, false, nullptr, 0, nullptr, nullptr, false, wave, lane);
  stage_sync(wave);
  // ctx = relu((X + (o@Wp + bp)) * s + t) -> Xe
  gemm_stage<false>(Kb, LDH, w1, 128, 128, 128, Xe, LDH, Cf + obP, false, Xe, LDH, Cf + oS, Cf + oT, true, wave, lane);
  __syncthreads();
}

// ======================= fused network: one workgroup per batch element ================
__launch_bounds__(256)
__global__ void pcn_fused(const float* __restrict__ x,
                          const _Float16* __restrict__ wh,
                          const float* __restrict__ wf,
                          float* __restrict__ out) {
  __shared__ _Float16 Xe[128 * LDH];
  __shared__ _Float16 Kb[128 * LDH];
  __shared__ _Float16 Qb[128 * LDH];
  __shared__ _Float16 Ab[128 * LDH];
  __shared__ _Float16 WbA[128 * 128];   // weight double buffer (TDM destination)
  __shared__ _Float16 WbB[128 * 128];
  __shared__ float    Cf[F_TOTAL];      // folded biases / BN scale+shift

  const int tid  = threadIdx.x;
  const int lane = tid & 31;
  const int wave = tid >> 5;
  const int b    = blockIdx.x;

  // ---- zero-pad x staging region ----
  for (int i = tid; i < 128 * 32; i += 256) Ab[i] = (_Float16)0.0f;
  __syncthreads();
  const float* xb = x + (size_t)b * 256;
  { const int tok = tid >> 1, d = tid & 1; Ab[tok * 32 + d] = (_Float16)xb[tid]; }
  for (int i = tid; i < F_TOTAL; i += 256) Cf[i] = wf[i];
  stage_load(wh + H_W11, 32, 128, WbA, tid, wave);   // initial fill
  stage_sync(wave);

  // ---- encoder ----
  stage_load(wh + H_W12, 32, 32, WbB, tid, wave);
  gemm_stage<false>(Ab, 32, WbA, 128, 128, 32, Kb, LDH, Cf + F_B11, false, nullptr, 0, nullptr, nullptr, false, wave, lane); // h1_0
  stage_sync(wave);
  stage_load(wh + H_W13, 32, 32, WbA, tid, wave);
  gemm_stage<false>(Ab, 32, WbB, 32, 32, 32, Qb, LDH, Cf + F_B12, false, nullptr, 0, nullptr, nullptr, true, wave, lane);    // h1_1
  stage_sync(wave);
  stage_load(wh + H_W14, 32, 128, WbB, tid, wave);
  gemm_stage<false>(Qb, LDH, WbA, 32, 32, 32, Ab, LDH, Cf + F_B13, false, nullptr, 0, nullptr, nullptr, true, wave, lane);   // h1_2
  stage_sync(wave);
  stage_load(wh + H_R1K, 128, 128, WbA, tid, wave);
  // embedding = relu(h1_0 + relu(bn(conv1_4(h1_2))))
  gemm_stage<false>(Ab, LDH, WbB, 128, 128, 32, Xe, LDH, Cf + F_B14, true, Kb, LDH, nullptr, nullptr, true, wave, lane);
  stage_sync(wave);

  // ---- two relation blocks (k-weights already staged in WbA on entry) ----
  relation_block(Xe, Kb, Qb, Ab, WbA, WbB, Cf, wh, H_R1Q, H_R1V, H_R1P,
                 F_R1BK, F_R1BQ, F_R1BV, F_R1BP, F_S2, F_T2,
                 wh + H_R2K, 128, 128, tid, wave, lane);
  relation_block(Xe, Kb, Qb, Ab, WbA, WbB, Cf, wh, H_R2Q, H_R2V, H_R2P,
                 F_R2BK, F_R2BQ, F_R2BV, F_R2BP, F_S4, F_T4,
                 wh + H_W21, 128, 128, tid, wave, lane);

  // ---- decoder (W21 already staged in WbA) ----
  stage_load(wh + H_W22, 128, 32, WbB, tid, wave);
  gemm_stage<false>(Xe, LDH, WbA, 128, 128, 128, Kb, LDH, Cf + F_B21, false, nullptr, 0, nullptr, nullptr, false, wave, lane); // h2_0
  stage_sync(wave);
  stage_load(wh + H_W23, 32, 32, WbA, tid, wave);
  gemm_stage<false>(Kb, LDH, WbB, 32, 32, 128, Qb, LDH, Cf + F_B22, false, nullptr, 0, nullptr, nullptr, true, wave, lane);    // h2_1
  stage_sync(wave);
  stage_load(wh + H_W24, 32, 128, WbB, tid, wave);
  gemm_stage<false>(Qb, LDH, WbA, 32, 32, 32, Ab, LDH, Cf + F_B23, false, nullptr, 0, nullptr, nullptr, true, wave, lane);     // h2_2
  stage_sync(wave);
  stage_load(wh + H_WGEO, 128, 16, WbA, tid, wave);
  // decoded = relu(h2_0 + relu(bn(conv2_4(h2_2))))
  gemm_stage<false>(Ab, LDH, WbB, 128, 128, 32, Xe, LDH, Cf + F_B24, true, Kb, LDH, nullptr, nullptr, true, wave, lane);
  stage_sync(wave);

  // ---- geo head: [128x128] @ [128x16(pad)] -> sigmoid, 2 real columns ----
  {
    const int m0 = wave << 4;
    v8f acc = {};
    for (int k0 = 0; k0 < 128; k0 += 32) {
      v16h a  = load_a_frag(Xe, LDH, m0, k0, lane);
      v16h bb = load_b_frag(WbA, 16, k0, 0, lane);
      acc = __builtin_amdgcn_wmma_f32_16x16x32_f16(false, a, false, bb, (short)0, acc,
                                                   false, false);
    }
    const int n  = lane & 15;
    const int mb = m0 + ((lane >> 4) << 3);
    if (n < 2) {
      const float bias = Cf[F_BGEO + n];
#pragma unroll
      for (int j = 0; j < 8; ++j) {
        float d = acc[j] + bias;
        out[(size_t)b * 256 + (mb + j) * 2 + n] = 1.0f / (1.0f + __expf(-d));
      }
    }
  }
}

// ======================= host entry ====================================================
extern "C" void kernel_launch(void* const* d_in, const int* in_sizes, int n_in,
                              void* d_out, int out_size, void* d_ws, size_t ws_size,
                              hipStream_t stream) {
  static const int ins2canon[83] = {
      82,
      49, 48,  1, 0, 2, 3,   51, 50,  5, 4, 6, 7,
      53, 52,  9, 8, 10, 11, 55, 54, 13, 12, 14, 15,
      67, 66, 71, 70, 73, 72, 69, 68,
      75, 74, 79, 78, 81, 80, 77, 76,
      33, 32, 34, 35, 37, 36, 38, 39, 41, 40, 42, 43, 45, 44, 46, 47,
      57, 56, 17, 16, 18, 19, 59, 58, 21, 20, 22, 23,
      61, 60, 25, 24, 26, 27, 63, 62, 29, 28, 30, 31,
      65, 64};

  Ptrs P;
  for (int i = 0; i < 83; ++i) P.p[i] = nullptr;
  if (n_in >= 83) {
    const bool x_first = (in_sizes[0] == 1024 * 128 * 2);
    if (x_first) {
      for (int i = 0; i < 83; ++i) P.p[ins2canon[i]] = (const float*)d_in[i];
    } else {
      for (int i = 0; i < 83; ++i) P.p[i] = (const float*)d_in[i];
    }
  } else {
    for (int i = 0; i < n_in && i < 83; ++i) P.p[i] = (const float*)d_in[i];
  }

  _Float16* wh = (_Float16*)d_ws;
  float*    wf = (float*)((char*)d_ws + (size_t)H_TOTAL * sizeof(_Float16));
  (void)ws_size; (void)out_size;

  pcn_prep<<<1, 256, 0, stream>>>(P, wh, wf);
  pcn_fused<<<1024, 256, 0, stream>>>(P.p[82], wh, wf, (float*)d_out);
}
